// HookAttention_40527311405175
// MI455X (gfx1250) — matrix operand order repaired and checked by reading, not compiled, not run on hardware
//
#include <hip/hip_runtime.h>
#include <stdint.h>

// HookAttention prefill (causal, GQA 32q/8kv, D=128, B=8, S=1024).
// Paged scatter/gather is an identity for the output -> flash attention on q,k,v.

#define HQ_  32
#define HKV_ 8
#define D_   128
#define S_   1024
#define GQ_  4

typedef __attribute__((ext_vector_type(16))) __bf16 v16bf;
typedef __attribute__((ext_vector_type(8)))  float  v8f;

union BF16Frag {
  v16bf v;
  unsigned short u[16];
  unsigned e[8];
  uint4 q2[2];
};

// f32 -> bf16 via native clang cast (RNE); lowers inline on gfx1250
static __device__ __forceinline__ unsigned short f2bf(float f) {
  __bf16 h = (__bf16)f;
  unsigned short r;
  __builtin_memcpy(&r, &h, sizeof(r));
  return r;
}

// pack two f32 -> packed 2x bf16 (hw packed cvt when the builtin exists)
static __device__ __forceinline__ unsigned packbf2(float lo, float hi) {
#if __has_builtin(__builtin_amdgcn_cvt_pk_bf16_f32)
  auto t = __builtin_amdgcn_cvt_pk_bf16_f32(lo, hi);
  unsigned r;
  __builtin_memcpy(&r, &t, sizeof(r));
  return r;
#else
  return (unsigned)f2bf(lo) | ((unsigned)f2bf(hi) << 16);
#endif
}

static __device__ __forceinline__ void pack8(unsigned* e, float4 a, float4 b) {
  e[0] = packbf2(a.x, a.y);
  e[1] = packbf2(a.z, a.w);
  e[2] = packbf2(b.x, b.y);
  e[3] = packbf2(b.z, b.w);
}

static __device__ __forceinline__ float4 s4(float4 a, float s) {
  a.x *= s; a.y *= s; a.z *= s; a.w *= s;
  return a;
}

static __device__ __forceinline__ v8f wmma_bf16(v16bf a, v16bf b, v8f c) {
  return __builtin_amdgcn_wmma_f32_16x16x32_bf16(false, a, false, b, (short)0, c, false, false);
}

static __device__ __forceinline__ v8f z8() {
  v8f z = {0.f, 0.f, 0.f, 0.f, 0.f, 0.f, 0.f, 0.f};
  return z;
}

__global__ __launch_bounds__(256) void HookAttention_40527311405175_kernel(
    const float* __restrict__ q, const float* __restrict__ k,
    const float* __restrict__ v, float* __restrict__ out) {

  extern __shared__ char smem[];
  float*          Kst = (float*)smem;                      // [2][32*128] f32 async staging (32 KB)
  float*          Vst = (float*)(smem + 32768);            // [2][32*128] f32 async staging (32 KB)
  unsigned short* Kt  = (unsigned short*)(smem + 65536);   // [128][32] bf16, dim-major (8 KB)
  unsigned short* Vb  = (unsigned short*)(smem + 73728);   // [32][128] bf16, key-major (8 KB)
  unsigned short* Pl  = (unsigned short*)(smem + 81920);   // [8][16*32] bf16 P staging (8 KB)

  const int tid  = threadIdx.x;
  const int w    = tid >> 5;       // wave 0..7
  const int lane = tid & 31;
  const int n16  = lane & 15;      // N column within 16-wide tile
  const int kh   = lane >> 4;      // half-wave selector

  const int blk     = blockIdx.x;
  const int mChunk  = blk & 31;            // S/32 = 32 row chunks
  const int kvh     = (blk >> 5) & 7;
  const int b       = blk >> 8;
  const int mB      = mChunk * 32;
  const int qh      = kvh * GQ_ + (w & 3);
  const int rowBase = mB + 16 * (w >> 2);  // this wave's 16-row query tile

  const size_t    tokBase = (size_t)b * S_;
  unsigned short* Pw      = Pl + w * (16 * 32);

  const float* kBase = k + (tokBase * HKV_ + kvh) * D_;
  const float* vBase = v + (tokBase * HKV_ + kvh) * D_;

  const unsigned ldsK0 = (unsigned)(uintptr_t)Kst;
  const unsigned ldsV0 = (unsigned)(uintptr_t)Vst;

  const float sMul = 0.08838834764831845f * 1.4426950408889634f; // 1/sqrt(D)*log2(e)

  // ---- Q fragments: A-matrix layout, pre-scaled so scores land in exp2 domain ----
  BF16Frag qa[4];
  {
    const float* qrow = q + ((tokBase + (size_t)(rowBase + n16)) * HQ_ + qh) * D_;
    #pragma unroll
    for (int f = 0; f < 4; ++f) {
      const float* p = qrow + f * 32 + kh * 8;
      float4 c0 = s4(*(const float4*)(p),      sMul);
      float4 c1 = s4(*(const float4*)(p + 4),  sMul);
      float4 c2 = s4(*(const float4*)(p + 16), sMul);
      float4 c3 = s4(*(const float4*)(p + 20), sMul);
      pack8(&qa[f].e[0], c0, c1);
      pack8(&qa[f].e[4], c2, c3);
    }
  }

  // ones B-fragment for WMMA row-sum
  BF16Frag onesF;
  #pragma unroll
  for (int i = 0; i < 8; ++i) onesF.e[i] = 0x3F803F80u;

  v8f acc[8];
  #pragma unroll
  for (int dt = 0; dt < 8; ++dt) acc[dt] = z8();

  float mx[8], ls[8];
  #pragma unroll
  for (int r = 0; r < 8; ++r) { mx[r] = -1e30f; ls[r] = 0.0f; }

  const int ntiles = (mB >> 5) + 1;   // causal: keys up to mB+31

  // async global->LDS DMA of one 32-key x 128-dim f32 K and V tile (16 KB each)
  auto issueTile = [&](int j, int bufSel) {
    const unsigned ldsK = ldsK0 + (unsigned)(bufSel * 16384);
    const unsigned ldsV = ldsV0 + (unsigned)(bufSel * 16384);
    #pragma unroll
    for (int i = 0; i < 4; ++i) {
      int id  = tid + 256 * i;
      int key = id >> 5;          // 32 b128-chunks per 512-byte key row
      int cir = id & 31;
      unsigned goff = (unsigned)(((j * 32 + key) * (HKV_ * D_) + cir * 4) * 4);
      unsigned loK  = ldsK + (unsigned)((key * D_ + cir * 4) * 4);
      unsigned loV  = ldsV + (unsigned)((key * D_ + cir * 4) * 4);
      asm volatile("global_load_async_to_lds_b128 %0, %1, %2"
                   :: "v"(loK), "v"(goff), "s"(kBase) : "memory");
      asm volatile("global_load_async_to_lds_b128 %0, %1, %2"
                   :: "v"(loV), "v"(goff), "s"(vBase) : "memory");
    }
  };

  // one key-tile step; `masked`/`issueNext` are call-site constants (cloned bodies)
  auto processTile = [&](int j, bool masked, bool issueNext) {
    const int bufSel = j & 1;
    asm volatile("s_wait_asynccnt 0" ::: "memory");
    __syncthreads();                                    // staging[bufSel] visible to all
    if (issueNext) issueTile(j + 1, bufSel ^ 1);        // overlap next tile's DMA

    const float* Ks = Kst + bufSel * (32 * 128);
    const float* Vs = Vst + bufSel * (32 * 128);

    // ---- cooperative one-shot f32 -> bf16 tile conversion ----
    {
      // V: key-major copy. thread -> (key = tid/8, dim segment = (tid%8)*16)
      const float*    vsrc = Vs + (tid >> 3) * D_ + (tid & 7) * 16;
      unsigned short* vdst = Vb + (tid >> 3) * D_ + (tid & 7) * 16;
      float4 a0 = *(const float4*)(vsrc);
      float4 a1 = *(const float4*)(vsrc + 4);
      float4 a2 = *(const float4*)(vsrc + 8);
      float4 a3 = *(const float4*)(vsrc + 12);
      BF16Frag tv;
      pack8(&tv.e[0], a0, a1);
      pack8(&tv.e[4], a2, a3);
      *(uint4*)(vdst)     = tv.q2[0];
      *(uint4*)(vdst + 8) = tv.q2[1];

      // K: transpose to dim-major. thread -> (dim = tid&127, key segment = (tid>>7)*16)
      const int dim = tid & 127, k0 = (tid >> 7) * 16;
      const float*    ksrc = Ks + k0 * D_ + dim;
      unsigned short* kdst = Kt + dim * 32 + k0;
      BF16Frag tk;
      #pragma unroll
      for (int kk = 0; kk < 16; kk += 2)
        tk.e[kk >> 1] = packbf2(ksrc[kk * D_], ksrc[(kk + 1) * D_]);
      *(uint4*)(kdst)     = tk.q2[0];
      *(uint4*)(kdst + 8) = tk.q2[1];
    }
    __syncthreads();                                    // Kt/Vb ready

    // ---- scores S = Q * K^T (already scaled): two 16-key tiles, 4 d-slices ----
    v8f sc0 = z8(), sc1 = z8();
    #pragma unroll
    for (int f = 0; f < 4; ++f) {
      const unsigned short* kp = Kt + (f * 32 + kh * 8) * 32 + n16;
      BF16Frag kb0, kb1;
      #pragma unroll
      for (int s = 0; s < 16; ++s) {
        const int d = (s < 8) ? s : s + 8;
        kb0.u[s] = kp[d * 32];
        kb1.u[s] = kp[d * 32 + 16];
      }
      sc0 = wmma_bf16(qa[f].v, kb0.v, sc0);
      sc1 = wmma_bf16(qa[f].v, kb1.v, sc1);
    }

    // ---- online softmax: row-max via 4-step butterfly; P staged for C->A relayout ----
    const int keyBase = j * 32;
    float cor[8];
    #pragma unroll
    for (int r = 0; r < 8; ++r) {
      float s0 = sc0[r];
      float s1 = sc1[r];
      if (masked) {
        const int mrow = rowBase + r + 8 * kh;  // C layout: half-waves own rows r / r+8
        if (keyBase + n16 > mrow)      s0 = -1e30f;
        if (keyBase + 16 + n16 > mrow) s1 = -1e30f;
      }
      float tmax = fmaxf(s0, s1);
      #pragma unroll
      for (int off = 8; off > 0; off >>= 1)
        tmax = fmaxf(tmax, __shfl_xor(tmax, off, 16));
      const float mnew = fmaxf(mx[r], tmax);
      const float p0 = exp2f(s0 - mnew);
      const float p1 = exp2f(s1 - mnew);
      cor[r] = exp2f(mx[r] - mnew);
      mx[r]  = mnew;
      #pragma unroll
      for (int dt = 0; dt < 8; ++dt) acc[dt][r] *= cor[r];
      Pw[(r + 8 * kh) * 32 + n16]      = f2bf(p0);
      Pw[(r + 8 * kh) * 32 + 16 + n16] = f2bf(p1);
    }

    // ---- P fragment in A-matrix layout (same-wave LDS bounce, DS in-order) ----
    BF16Frag pa;
    pa.q2[0] = *(const uint4*)(Pw + n16 * 32 + kh * 8);
    pa.q2[1] = *(const uint4*)(Pw + n16 * 32 + 16 + kh * 8);

    // ---- row sums with one WMMA (P x ones) instead of a shuffle tree ----
    {
      v8f rsv = wmma_bf16(pa.v, onesF.v, z8());
      #pragma unroll
      for (int r = 0; r < 8; ++r) ls[r] = ls[r] * cor[r] + rsv[r];
    }

    // ---- O += P * V : K=32 in one WMMA, 8 dim tiles of N=16 ----
    #pragma unroll
    for (int dt = 0; dt < 8; ++dt) {
      const unsigned short* vp = Vb + (kh * 8) * D_ + dt * 16 + n16;
      BF16Frag vb;
      #pragma unroll
      for (int s = 0; s < 16; ++s) {
        const int d = (s < 8) ? s : s + 8;
        vb.u[s] = vp[d * D_];
      }
      acc[dt] = wmma_bf16(pa.v, vb.v, acc[dt]);
    }
    // no trailing barrier: next iteration's entry barrier orders
    // compute-reads of Kt/Vb against the next conversion pass.
  };

  issueTile(0, 0);
  for (int j = 0; j + 1 < ntiles; ++j)
    processTile(j, /*masked=*/false, /*issueNext=*/true);   // fully-valid tiles
  processTile(ntiles - 1, /*masked=*/true, /*issueNext=*/false); // diagonal tile

  // ---- epilogue: normalize and store 16x128 f32 tile ----
  #pragma unroll
  for (int r = 0; r < 8; ++r) {
    const float  invl = 1.0f / ls[r];
    const size_t base =
        ((tokBase + (size_t)(rowBase + r + 8 * kh)) * HQ_ + qh) * D_ + n16;
    #pragma unroll
    for (int dt = 0; dt < 8; ++dt)
      out[base + dt * 16] = acc[dt][r] * invl;
  }
}

extern "C" void kernel_launch(void* const* d_in, const int* in_sizes, int n_in,
                              void* d_out, int out_size, void* d_ws, size_t ws_size,
                              hipStream_t stream) {
  (void)in_sizes; (void)n_in; (void)out_size; (void)d_ws; (void)ws_size;
  const float* q = (const float*)d_in[0];
  const float* k = (const float*)d_in[1];
  const float* v = (const float*)d_in[2];
  float* out = (float*)d_out;

  const int smemBytes = 32768 + 32768 + 8192 + 8192 + 8192;  // 90112 B dynamic LDS
  hipFuncSetAttribute(
      reinterpret_cast<const void*>(&HookAttention_40527311405175_kernel),
      hipFuncAttributeMaxDynamicSharedMemorySize, smemBytes);

  // grid: B * HKV * (S/32) = 8 * 8 * 32 = 2048 blocks, 256 threads (8 wave32)
  HookAttention_40527311405175_kernel<<<dim3(8 * 8 * 32), dim3(256), smemBytes,
                                        stream>>>(q, k, v, out);
}
